// ApcaNetContaper_6090263625905
// MI455X (gfx1250) — matrix operand order repaired and checked by reading, not compiled, not run on hardware
//
#include <hip/hip_runtime.h>

// ---------------- types for WMMA ----------------
typedef __attribute__((ext_vector_type(16))) __bf16 v16bf;
typedef __attribute__((ext_vector_type(8)))  float  v8f;

#define F_TRANS_B  1
#define F_TRANS_C  2
#define F_BIAS     4
#define F_RELU     8
#define F_MISH     16
#define F_RES_ROW  32   // R[m*ldr + n]
#define F_RES_COL  64   // R[n*ldr + m]

__device__ __forceinline__ unsigned short f2bf(float f) {
  unsigned u = __float_as_uint(f);
  unsigned r = u + 0x7FFFu + ((u >> 16) & 1u);   // round-to-nearest-even
  return (unsigned short)(r >> 16);
}

__device__ __forceinline__ float mishf(float x) {
  float sp = (x > 20.f) ? x : log1pf(__expf(x));
  return x * tanhf(sp);
}

// k-position permutation so each lane's A-fragment data is contiguous in LDS:
// WMMA 16-bit A layout: lane(hi,row) element e holds k=(e&7)+((e>>3)<<4)+hi*8
// -> store k at position p = swap(bit3,bit4)(k); lane reads [hi*16, hi*16+16).
// Note: within any aligned 8-group of k, bits 3/4 are constant, so a 16-byte
// chunk stays contiguous after permutation (async-DMA friendly).
__device__ __forceinline__ int permk(int c) {
  return (c & 7) | ((c >> 1) & 8) | ((c << 1) & 16);
}

// ---------------- batched WMMA GEMM: C[M,N] = act(A[M,K] * B[K,N] + bias + res) ----
// A: bf16 row-major (pre-converted). B: f32, normal B[k*ldb+n] or TRANS_B B[n*ldb+k].
// C store: normal C[m*ldc+n] or TRANS_C C[n*ldc+m].
// Batch z = n*H + h with independent strides per n and per h for each operand.
__global__ __launch_bounds__(128) void gemm_wmma_bf16(
    const unsigned short* __restrict__ A, const float* __restrict__ B,
    float* __restrict__ C, const float* __restrict__ R,
    const float* __restrict__ bias,
    int M, int Nn, int K,
    int lda, int ldb, int ldc, int ldr,
    long sAn, long sBn, long sCn, long sRn,
    int H, long sAh, long sBh, long sCh, long sRh,
    int flags)
{
  int z = blockIdx.z;
  int bn = z / H, bh = z - bn * H;
  A += (long)bn * sAn + (long)bh * sAh;
  B += (long)bn * sBn + (long)bh * sBh;
  C += (long)bn * sCn + (long)bh * sCh;
  if (R) R += (long)bn * sRn + (long)bh * sRh;

  __shared__ unsigned short Ap[64][32];   // [m][perm(k)]  fragment-ready
  __shared__ unsigned short Bt[64][32];   // [n][k]        fragment-ready (transposed)

  const int tid  = threadIdx.x;
  const int wave = tid >> 5;
  const int lane = tid & 31;
  const int row16 = lane & 15;
  const int hi    = lane >> 4;

  const int m0 = blockIdx.y * 64;
  const int n0 = blockIdx.x * 64;
  const bool fullM = (m0 + 64 <= M);
  const bool fullN = (n0 + 64 <= Nn);

  v8f acc[4] = {};

  for (int k0 = 0; k0 < K; k0 += 32) {
    const bool fullK = (k0 + 32 <= K);

    // ---- stage A tile (64x32 bf16): async DMA global -> LDS, k-permuted ----
    bool asyncA = fullM && fullK;
    if (asyncA) {
      #pragma unroll
      for (int t = tid; t < 256; t += 128) {       // 256 chunks of 16B, 2 per thread
        int r = t >> 2, cg = (t & 3) << 3;         // cg in {0,8,16,24}
        const unsigned short* gp = A + (long)(m0 + r) * lda + (k0 + cg);
        unsigned lds = (unsigned)(unsigned long long)(&Ap[r][permk(cg)]);
        asm volatile("global_load_async_to_lds_b128 %0, %1, off"
                     :: "v"(lds), "v"(gp) : "memory");
      }
    } else {
      #pragma unroll
      for (int i = tid; i < 2048; i += 128) {
        int r = i >> 5, c = i & 31;
        int m = m0 + r, k = k0 + c;
        int mc = m < M ? m : M - 1;
        int kc = k < K ? k : K - 1;
        unsigned short v = A[(long)mc * lda + kc];  // always-valid address
        v = (m < M && k < K) ? v : (unsigned short)0;
        Ap[r][permk(c)] = v;
      }
    }

    // ---- stage B tile (32x64) f32 -> bf16, transposed into Bt[n][k] ----
    if (fullN && fullK) {
      if (flags & F_TRANS_B) {
        #pragma unroll
        for (int t = tid; t < 512; t += 128) {
          int c = t & 63, kg = (t >> 6) << 2;
          const float4 v = *(const float4*)(B + (long)(n0 + c) * ldb + (k0 + kg));
          ushort4 s;
          s.x = f2bf(v.x); s.y = f2bf(v.y); s.z = f2bf(v.z); s.w = f2bf(v.w);
          *(ushort4*)(&Bt[c][kg]) = s;
        }
      } else {
        #pragma unroll
        for (int t = tid; t < 512; t += 128) {
          int c = t & 63, kg = (t >> 6) << 2;
          const float* bp = B + (long)(k0 + kg) * ldb + (n0 + c);
          ushort4 s;
          s.x = f2bf(bp[0]);
          s.y = f2bf(bp[ldb]);
          s.z = f2bf(bp[2 * (long)ldb]);
          s.w = f2bf(bp[3 * (long)ldb]);
          *(ushort4*)(&Bt[c][kg]) = s;
        }
      }
    } else {
      #pragma unroll
      for (int i = tid; i < 2048; i += 128) {
        int c = i >> 5, r = i & 31;
        int k = k0 + r, nn = n0 + c;
        int kc = k < K ? k : K - 1;
        int nc = nn < Nn ? nn : Nn - 1;
        long idx = (flags & F_TRANS_B) ? ((long)nc * ldb + kc) : ((long)kc * ldb + nc);
        float v = B[idx];
        v = (k < K && nn < Nn) ? v : 0.f;
        Bt[c][r] = f2bf(v);
      }
    }

    // drain this wave's async DMAs before the workgroup barrier
    if (asyncA) {
      asm volatile("s_wait_asynccnt 0" ::: "memory");
    }
    __syncthreads();

    // ---- fragments: 2x b128 LDS loads each ----
    union Frag { uint4 q[2]; v16bf v; };
    Frag fa;
    {
      const unsigned short* ap = &Ap[wave * 16 + row16][hi * 16];
      fa.q[0] = *(const uint4*)(ap);
      fa.q[1] = *(const uint4*)(ap + 8);
    }
    #pragma unroll
    for (int f = 0; f < 4; f++) {
      Frag fb;
      const unsigned short* bp = &Bt[f * 16 + row16][hi * 8];
      fb.q[0] = *(const uint4*)(bp);
      fb.q[1] = *(const uint4*)(bp + 16);
      acc[f] = __builtin_amdgcn_wmma_f32_16x16x32_bf16(
          false, fa.v, false, fb.v, (short)0, acc[f], false, false);
    }
    __syncthreads();
  }

  // epilogue: bias / residual / activation / (transposed) store
  #pragma unroll
  for (int f = 0; f < 4; f++) {
    union CF { float x[8]; v8f v; } cf;
    cf.v = acc[f];
    int nn = n0 + f * 16 + row16;
    #pragma unroll
    for (int r = 0; r < 8; r++) {
      int m = m0 + wave * 16 + hi * 8 + r;
      if (m < M && nn < Nn) {
        float v = cf.x[r];
        if (flags & F_BIAS)    v += bias[m];
        if (flags & F_RES_ROW) v += R[(long)m * ldr + nn];
        if (flags & F_RES_COL) v += R[(long)nn * ldr + m];
        if (flags & F_RELU)    v = fmaxf(v, 0.f);
        if (flags & F_MISH)    v = mishf(v);
        if (flags & F_TRANS_C) C[(long)nn * ldc + m] = v;
        else                   C[(long)m * ldc + nn] = v;
      }
    }
  }
}

// ---------------- f32 -> bf16 elementwise convert ----------------
__global__ __launch_bounds__(256) void to_bf16(
    const float* __restrict__ src, unsigned short* __restrict__ dst, long n)
{
  long i = (long)blockIdx.x * 256 + threadIdx.x;
  if (i < n) dst[i] = f2bf(src[i]);
}

// ---------------- grouped conv1d (small groups) ----------------
__global__ __launch_bounds__(256) void grouped_conv1d(
    const float* __restrict__ in, const float* __restrict__ w,
    const float* __restrict__ bias, float* __restrict__ out,
    int Cout, int L, int gin, int gout, int ksz, int pad, int act,
    long sInN, long sOutN)
{
  long idx = (long)blockIdx.x * blockDim.x + threadIdx.x;
  int n = blockIdx.y;
  if (idx >= (long)Cout * L) return;
  int co = (int)(idx / L);
  int l  = (int)(idx % L);
  int g  = co / gout;
  const float* ip = in + (long)n * sInN + (long)g * gin * L;
  const float* wp = w + (long)co * gin * ksz;
  float acc = bias[co];
  for (int ic = 0; ic < gin; ic++) {
    for (int t = 0; t < ksz; t++) {
      int p = l + t - pad;
      if (p >= 0 && p < L) acc += wp[ic * ksz + t] * ip[(long)ic * L + p];
    }
  }
  if (act == 1) acc = mishf(acc);
  out[(long)n * sOutN + idx] = acc;
}

// ---------------- LayerNorm over channel dim (stride-L), optional pos add ----
__global__ __launch_bounds__(256) void ln_chan(
    const float* __restrict__ x, const float* __restrict__ g,
    const float* __restrict__ b, const float* __restrict__ pos,
    float* __restrict__ out, int C, int L, long sN)
{
  int n = blockIdx.y, l = blockIdx.x;
  const float* xp = x + (long)n * sN + l;
  __shared__ float s1[256], s2[256];
  float sum = 0.f, sq = 0.f;
  for (int c = threadIdx.x; c < C; c += 256) {
    float v = xp[(long)c * L]; sum += v; sq += v * v;
  }
  s1[threadIdx.x] = sum; s2[threadIdx.x] = sq; __syncthreads();
  for (int s = 128; s > 0; s >>= 1) {
    if ((int)threadIdx.x < s) { s1[threadIdx.x] += s1[threadIdx.x + s]; s2[threadIdx.x] += s2[threadIdx.x + s]; }
    __syncthreads();
  }
  float mu = s1[0] / C;
  float var = s2[0] / C - mu * mu;
  float rs = rsqrtf(var + 1e-5f);
  float* op = out + (long)n * sN + l;
  for (int c = threadIdx.x; c < C; c += 256) {
    float v = (xp[(long)c * L] - mu) * rs * g[c] + b[c];
    if (pos) v += pos[(long)c * L + l];
    op[(long)c * L] = v;
  }
}

// ---------------- LayerNorm over contiguous rows, optional transposed output ----
__global__ __launch_bounds__(256) void ln_rows(
    const float* __restrict__ x, const float* __restrict__ g,
    const float* __restrict__ b, float* __restrict__ out,
    int C, int L, int transOut)
{
  long row = blockIdx.x;
  const float* xp = x + row * (long)C;
  __shared__ float s1[256], s2[256];
  float sum = 0.f, sq = 0.f;
  for (int c = threadIdx.x; c < C; c += 256) { float v = xp[c]; sum += v; sq += v * v; }
  s1[threadIdx.x] = sum; s2[threadIdx.x] = sq; __syncthreads();
  for (int s = 128; s > 0; s >>= 1) {
    if ((int)threadIdx.x < s) { s1[threadIdx.x] += s1[threadIdx.x + s]; s2[threadIdx.x] += s2[threadIdx.x + s]; }
    __syncthreads();
  }
  float mu = s1[0] / C;
  float var = s2[0] / C - mu * mu;
  float rs = rsqrtf(var + 1e-5f);
  long n = row / L, l = row % L;
  for (int c = threadIdx.x; c < C; c += 256) {
    float v = (xp[c] - mu) * rs * g[c] + b[c];
    if (transOut) out[(n * C + c) * L + l] = v;
    else          out[row * (long)C + c] = v;
  }
}

// ---------------- softmax rows with pre-scale ----------------
__global__ __launch_bounds__(256) void softmax_rows(float* __restrict__ x, int C, float scale)
{
  long row = blockIdx.x;
  float* xp = x + row * (long)C;
  __shared__ float red[256];
  float m = -3.4e38f;
  for (int i = threadIdx.x; i < C; i += 256) m = fmaxf(m, xp[i] * scale);
  red[threadIdx.x] = m; __syncthreads();
  for (int s = 128; s > 0; s >>= 1) {
    if ((int)threadIdx.x < s) red[threadIdx.x] = fmaxf(red[threadIdx.x], red[threadIdx.x + s]);
    __syncthreads();
  }
  m = red[0]; __syncthreads();
  float sum = 0.f;
  for (int i = threadIdx.x; i < C; i += 256) {
    float e = __expf(xp[i] * scale - m); xp[i] = e; sum += e;
  }
  red[threadIdx.x] = sum; __syncthreads();
  for (int s = 128; s > 0; s >>= 1) {
    if ((int)threadIdx.x < s) red[threadIdx.x] += red[threadIdx.x + s];
    __syncthreads();
  }
  float inv = 1.f / red[0];
  for (int i = threadIdx.x; i < C; i += 256) xp[i] *= inv;
}

// ---------------- final LN + reshape (n,12288,200)->(n,512,4800) + residual ----
__global__ __launch_bounds__(256) void ln_final(
    const float* __restrict__ x, const float* __restrict__ g,
    const float* __restrict__ b, const float* __restrict__ res,
    float* __restrict__ out, int C, int L)
{
  int n = blockIdx.y, l = blockIdx.x;
  long sN = (long)C * L;
  const float* xp = x + (long)n * sN + l;
  __shared__ float s1[256], s2[256];
  float sum = 0.f, sq = 0.f;
  for (int c = threadIdx.x; c < C; c += 256) {
    float v = xp[(long)c * L]; sum += v; sq += v * v;
  }
  s1[threadIdx.x] = sum; s2[threadIdx.x] = sq; __syncthreads();
  for (int s = 128; s > 0; s >>= 1) {
    if ((int)threadIdx.x < s) { s1[threadIdx.x] += s1[threadIdx.x + s]; s2[threadIdx.x] += s2[threadIdx.x + s]; }
    __syncthreads();
  }
  float mu = s1[0] / C;
  float var = s2[0] / C - mu * mu;
  float rs = rsqrtf(var + 1e-5f);
  for (int c = threadIdx.x; c < C; c += 256) {
    float v = (xp[(long)c * L] - mu) * rs * g[c] + b[c];
    int c0 = c / 24, j = c % 24;             // (n,24*D,L) -> (n,D,24*L)
    long oi = ((long)n * 512 + c0) * 4800 + (long)j * L + l;
    out[oi] = v + res[oi];
  }
}

// ---------------- host-side launch helpers ----------------
static inline void launch_gemm(hipStream_t s,
    const unsigned short* A, const float* B, float* C, const float* R, const float* bias,
    int M, int Nn, int K, int lda, int ldb, int ldc, int ldr,
    long sAn, long sBn, long sCn, long sRn,
    int Nb, int H, long sAh, long sBh, long sCh, long sRh, int flags)
{
  dim3 grid((unsigned)((Nn + 63) / 64), (unsigned)((M + 63) / 64), (unsigned)(Nb * H));
  gemm_wmma_bf16<<<grid, 128, 0, s>>>(A, B, C, R, bias, M, Nn, K,
      lda, ldb, ldc, ldr, sAn, sBn, sCn, sRn, H, sAh, sBh, sCh, sRh, flags);
}

extern "C" void kernel_launch(void* const* d_in, const int* in_sizes, int n_in,
                              void* d_out, int out_size, void* d_ws, size_t ws_size,
                              hipStream_t stream) {
  (void)in_sizes; (void)n_in; (void)out_size; (void)ws_size;
  const int NB = 8, DCH = 512, TT = 4800, LL = 200;
  const int C6 = 3072, C24 = 12288, EE = 1024, HH = 8, HD = 128, FFW = 2048;

  const float* x    = (const float*)d_in[0];
  const float* cond = (const float*)d_in[1];
  const float* diff = (const float*)d_in[2];
  int i = 3;
  const float* w_diff = (const float*)d_in[i++];
  const float* b_diff = (const float*)d_in[i++];
  const float* w_cond = (const float*)d_in[i++];
  const float* b_cond = (const float*)d_in[i++];
  const float* w_v1   = (const float*)d_in[i++];
  const float* b_v1   = (const float*)d_in[i++];
  const float* w_v2   = (const float*)d_in[i++];
  const float* b_v2   = (const float*)d_in[i++];
  const float* g_v    = (const float*)d_in[i++];
  const float* bt_v   = (const float*)d_in[i++];
  const float* w_v3   = (const float*)d_in[i++];
  const float* b_v3   = (const float*)d_in[i++];
  const float* Wq     = (const float*)d_in[i++];
  const float* Wk     = (const float*)d_in[i++];
  const float* Wv     = (const float*)d_in[i++];
  const float* Wo     = (const float*)d_in[i++];
  const float* bo     = (const float*)d_in[i++];
  const float* g1     = (const float*)d_in[i++];
  const float* bt1    = (const float*)d_in[i++];
  const float* g2     = (const float*)d_in[i++];
  const float* bt2    = (const float*)d_in[i++];
  const float* Wf1    = (const float*)d_in[i++];
  const float* bf1    = (const float*)d_in[i++];
  const float* Wf2    = (const float*)d_in[i++];
  const float* bf2    = (const float*)d_in[i++];
  const float* w_p1   = (const float*)d_in[i++];
  const float* b_p1   = (const float*)d_in[i++];
  const float* w_p2   = (const float*)d_in[i++];
  const float* b_p2   = (const float*)d_in[i++];
  const float* g_p    = (const float*)d_in[i++];
  const float* bt_p   = (const float*)d_in[i++];
  const float* pos    = (const float*)d_in[i++];

  // workspace carve-up (f32 region)
  float* p = (float*)d_ws;
  float* y0   = p; p += (long)NB * DCH * TT;       // (n,512,4800) == (n,12288,200)
  float* bufA = p; p += (long)NB * C6 * LL;        // v1 out
  float* bufB = p; p += (long)NB * C6 * LL;        // v2 / ln+pos
  float* qkv  = p; p += (long)NB * C6 * LL;        // v3 out
  float* qh   = p; p += (long)NB * LL * EE;        // (n,L,E)
  float* kh   = p; p += (long)NB * EE * LL;        // (n,h,d,L)
  float* vh   = p; p += (long)NB * LL * EE;        // (n,h,L,d)
  float* att  = p; p += (long)NB * HH * LL * LL;   // (n,h,L,L)
  float* ao   = p; p += (long)NB * LL * EE;        // (n,L,E)
  float* x1   = p; p += (long)NB * LL * EE;        // (n,L,E)
  float* ff1  = p; p += (long)NB * FFW * LL;       // (n,2E,L)
  float* le   = p; p += (long)NB * LL * EE;        // (n,L,E)
  float* x2   = p; p += (long)NB * EE * LL;        // (n,E,L)
  float* p1b  = p; p += (long)NB * C6 * LL;        // (n,3072,200)
  float* p2b  = p; p += (long)NB * C24 * LL;       // (n,12288,200)

  // bf16 region (A operands, pre-converted; all sizes multiples of 8 -> 16B aligned)
  unsigned short* ub = (unsigned short*)p;
  unsigned short* wdiff_h = ub; ub += (long)DCH * DCH;   // 512x512
  unsigned short* wcond_h = ub; ub += (long)DCH * 256;   // 512x256
  unsigned short* wv3_h   = ub; ub += (long)C6 * C6;     // 3072x3072
  unsigned short* Wq_h    = ub; ub += (long)HD * HD;
  unsigned short* Wk_h    = ub; ub += (long)HD * HD;
  unsigned short* Wv_h    = ub; ub += (long)HD * HD;
  unsigned short* Wo_h    = ub; ub += (long)EE * EE;
  unsigned short* Wf1_h   = ub; ub += (long)FFW * EE;
  unsigned short* Wf2_h   = ub; ub += (long)EE * FFW;
  unsigned short* wp1_h   = ub; ub += (long)C6 * EE;
  unsigned short* qh_h    = ub; ub += (long)NB * LL * EE;
  unsigned short* att_h   = ub; ub += (long)NB * HH * LL * LL;

  const long sX   = (long)DCH * TT;     // 2457600
  const long sC6  = (long)C6 * LL;      // 614400
  const long sE   = (long)EE * LL;      // 204800 (== L*E)
  const long sFF  = (long)FFW * LL;     // 409600
  const long sATT = (long)HH * LL * LL; // 320000

  auto conv = [&](const float* s, unsigned short* d, long n) {
    to_bf16<<<dim3((unsigned)((n + 255) / 256)), 256, 0, stream>>>(s, d, n);
  };

  // pre-convert every GEMM A operand that is launch-invariant (weights)
  conv(w_diff, wdiff_h, (long)DCH * DCH);
  conv(w_cond, wcond_h, (long)DCH * 256);
  conv(w_v3,   wv3_h,   (long)C6 * C6);
  conv(Wq,     Wq_h,    (long)HD * HD);
  conv(Wk,     Wk_h,    (long)HD * HD);
  conv(Wv,     Wv_h,    (long)HD * HD);
  conv(Wo,     Wo_h,    (long)EE * EE);
  conv(Wf1,    Wf1_h,   (long)FFW * EE);
  conv(Wf2,    Wf2_h,   (long)EE * FFW);
  conv(w_p1,   wp1_h,   (long)C6 * EE);

  // 1) y0 = x + W_diff @ diffstep + b_diff
  launch_gemm(stream, wdiff_h, diff, y0, x, b_diff,
      DCH, TT, DCH, DCH, TT, TT, TT,
      0, sX, sX, sX, NB, 1, 0, 0, 0, 0, F_BIAS | F_RES_ROW);
  // 2) y0 += W_cond @ cond + b_cond
  launch_gemm(stream, wcond_h, cond, y0, y0, b_cond,
      DCH, TT, 256, 256, TT, TT, TT,
      0, (long)256 * TT, sX, sX, NB, 1, 0, 0, 0, 0, F_BIAS | F_RES_ROW);
  // 3) v1 grouped conv (12->3, k=1) + mish
  {
    dim3 g((unsigned)((sC6 + 255) / 256), NB);
    grouped_conv1d<<<g, 256, 0, stream>>>(y0, w_v1, b_v1, bufA, C6, LL, 12, 3, 1, 0, 1, sX, sC6);
  }
  // 4) v2 grouped conv (2->2, k=81, pad=40)
  {
    dim3 g((unsigned)((sC6 + 255) / 256), NB);
    grouped_conv1d<<<g, 256, 0, stream>>>(bufA, w_v2, b_v2, bufB, C6, LL, 2, 2, 81, 40, 0, sC6, sC6);
  }
  // 5) LN over channels + pos add
  ln_chan<<<dim3(LL, NB), 256, 0, stream>>>(bufB, g_v, bt_v, pos, bufB, C6, LL, sC6);
  // 6) v3: qkv = w_v3 @ x + b_v3
  launch_gemm(stream, wv3_h, bufB, qkv, nullptr, b_v3,
      C6, LL, C6, C6, LL, LL, 0,
      0, sC6, sC6, 0, NB, 1, 0, 0, 0, 0, F_BIAS);
  // 7) vh = Wv @ chunk0  -> (n,h,L,d)
  launch_gemm(stream, Wv_h, qkv, vh, nullptr, nullptr,
      HD, LL, HD, HD, LL, HD, 0,
      0, sC6, sE, 0, NB, HH, 0, (long)HD * LL, (long)LL * HD, 0, F_TRANS_C);
  // 8) kh = Wk @ chunk1  -> (n,h,d,L)
  launch_gemm(stream, Wk_h, qkv + (long)EE * LL, kh, nullptr, nullptr,
      HD, LL, HD, HD, LL, LL, 0,
      0, sC6, sE, 0, NB, HH, 0, (long)HD * LL, (long)HD * LL, 0, 0);
  // 9) qh = Wq @ chunk2  -> (n,L,h*d)
  launch_gemm(stream, Wq_h, qkv + 2L * EE * LL, qh, nullptr, nullptr,
      HD, LL, HD, HD, LL, EE, 0,
      0, sC6, sE, 0, NB, HH, 0, (long)HD * LL, (long)HD, 0, F_TRANS_C);
  // qh -> bf16 (A operand of att GEMM)
  conv(qh, qh_h, (long)NB * LL * EE);
  // 10) att = qh @ kh  (L x L per head)
  launch_gemm(stream, qh_h, kh, att, nullptr, nullptr,
      LL, LL, HD, EE, LL, LL, 0,
      sE, sE, sATT, 0, NB, HH, (long)HD, (long)HD * LL, (long)LL * LL, 0, 0);
  // 11) softmax with 1/sqrt(E) scale
  softmax_rows<<<dim3((unsigned)(NB * HH * LL)), 256, 0, stream>>>(att, LL, 1.0f / 32.0f);
  // att -> bf16 (A operand of ao GEMM)
  conv(att, att_h, (long)NB * HH * LL * LL);
  // 12) ao = att @ vh -> (n,L,E)
  launch_gemm(stream, att_h, vh, ao, nullptr, nullptr,
      LL, HD, LL, LL, HD, EE, 0,
      sATT, sE, sE, 0, NB, HH, (long)LL * LL, (long)LL * HD, (long)HD, 0, 0);
  // 13) x1 = Wo @ ao^T + bo + qq  -> stored (n,L,E)
  launch_gemm(stream, Wo_h, ao, x1, qkv + 2L * EE * LL, bo,
      EE, LL, EE, EE, EE, EE, LL,
      0, sE, sE, sC6, NB, 1, 0, 0, 0, 0, F_BIAS | F_TRANS_B | F_TRANS_C | F_RES_ROW);
  // 14) LN1 (in place on rows of E)
  ln_rows<<<dim3((unsigned)(NB * LL)), 256, 0, stream>>>(x1, g1, bt1, x1, EE, LL, 0);
  // 15) ff1 = relu(Wf1 @ x1^T + bf1) -> (n,2E,L)
  launch_gemm(stream, Wf1_h, x1, ff1, nullptr, bf1,
      FFW, LL, EE, EE, EE, LL, 0,
      0, sE, sFF, 0, NB, 1, 0, 0, 0, 0, F_BIAS | F_TRANS_B | F_RELU);
  // 16) le = Wf2 @ ff1 + bf2 + x1 -> stored (n,L,E)
  launch_gemm(stream, Wf2_h, ff1, le, x1, bf2,
      EE, LL, FFW, FFW, LL, EE, EE,
      0, sFF, sE, sE, NB, 1, 0, 0, 0, 0, F_BIAS | F_TRANS_C | F_RES_COL);
  // 17) LN2 -> transposed out (n,E,L)
  ln_rows<<<dim3((unsigned)(NB * LL)), 256, 0, stream>>>(le, g2, bt2, x2, EE, LL, 1);
  // 18) p1 = mish(w_p1 @ x2 + b_p1) -> (n,3072,200)
  launch_gemm(stream, wp1_h, x2, p1b, nullptr, b_p1,
      C6, LL, EE, EE, LL, LL, 0,
      0, sE, sC6, 0, NB, 1, 0, 0, 0, 0, F_BIAS | F_MISH);
  // 19) p2 grouped conv (12->48, k=5, pad=2) -> (n,12288,200)
  {
    long tot = (long)C24 * LL;
    dim3 g((unsigned)((tot + 255) / 256), NB);
    grouped_conv1d<<<g, 256, 0, stream>>>(p1b, w_p2, b_p2, p2b, C24, LL, 12, 48, 5, 2, 0, sC6, (long)C24 * LL);
  }
  // 20) final LN + reshape + residual
  ln_final<<<dim3(LL, NB), 256, 0, stream>>>(p2b, g_p, bt_p, x, (float*)d_out, C24, LL);
}
